// MultiHeadedAttention_90683939488193
// MI455X (gfx1250) — matrix operand order repaired and checked by reading, not compiled, not run on hardware
//
#include <hip/hip_runtime.h>

// ---------------------------------------------------------------------------
// Multi-head attention for MI455X (gfx1250, wave32, WMMA bf16 + async-LDS)
// B=4, S=2048, D=1024, H=16, DH=64
// ---------------------------------------------------------------------------

#define N_B   4
#define N_S   2048
#define N_D   1024
#define N_H   16
#define N_DH  64
#define N_BS  (N_B * N_S)          // 8192

typedef __attribute__((ext_vector_type(16))) __bf16 v16bf;
typedef __attribute__((ext_vector_type(8)))  float  v8f;
typedef int v4i32 __attribute__((vector_size(16)));

union FragBF {
    v16bf v;
    uint4 q[2];
};

// ---------------- async global->LDS copy (CDNA5, ASYNCcnt-tracked) ----------
#if defined(__has_builtin)
#  if __has_builtin(__builtin_amdgcn_global_load_async_to_lds_b128)
#    define HAVE_ASYNC_COPY 1
#  endif
#endif
#ifndef HAVE_ASYNC_COPY
#  define HAVE_ASYNC_COPY 0
#endif

#if HAVE_ASYNC_COPY
__device__ __forceinline__ void async_copy_b128(const void* gsrc, void* ldst) {
    // Builtin signature (from clang diagnostics): pointers to 4xi32 vectors in
    // addrspace(1)/addrspace(3) + imm offset + imm cpol. A generic LDS pointer
    // carries the DS byte offset in its low 32 bits (flat aperture lives in
    // bits 63:32), so an integer round-trip yields the addrspace(3) pointer.
    __builtin_amdgcn_global_load_async_to_lds_b128(
        (__attribute__((address_space(1))) v4i32*)(unsigned long long)gsrc,
        (__attribute__((address_space(3))) v4i32*)(unsigned int)(unsigned long long)ldst,
        0, 0);
}
#  if __has_builtin(__builtin_amdgcn_s_wait_asynccnt)
#    define WAIT_ASYNC(n) __builtin_amdgcn_s_wait_asynccnt(n)
#  else
#    define WAIT_ASYNC(n) asm volatile("s_wait_asynccnt %0" :: "i"(n) : "memory")
#  endif
#else
#  define WAIT_ASYNC(n) ((void)0)
#endif

__device__ __forceinline__ unsigned short f2bf(float x) {
    unsigned int u = __float_as_uint(x);
    u += 0x7FFFu + ((u >> 16) & 1u);   // round-to-nearest-even
    return (unsigned short)(u >> 16);
}

// A-fragment (16x32 bf16, M x K): lane l<16 -> row l, K chunks {0..7, 16..23};
// lane l>=16 -> row l-16, K chunks {8..15, 24..31}.   (ISA 7.12.2)
__device__ __forceinline__ void loadA(const unsigned short* base, int stride,
                                      int lane, FragBF& f) {
    const unsigned short* p = base + (lane & 15) * stride + (lane >> 4) * 8;
    f.q[0] = *reinterpret_cast<const uint4*>(p);
    f.q[1] = *reinterpret_cast<const uint4*>(p + 16);
}

// B-fragment (32x16 bf16, K x N), column-major storage (column n has `stride`
// halves): lane l<16 -> col l, K chunks {0..7, 16..23}; lane l>=16 -> col, K chunks {8..15, 24..31}.
__device__ __forceinline__ void loadB(const unsigned short* base, int stride,
                                      int lane, FragBF& f) {
    const unsigned short* p = base + (lane & 15) * stride + (lane >> 4) * 8;
    f.q[0] = *reinterpret_cast<const uint4*>(p);
    f.q[1] = *reinterpret_cast<const uint4*>(p + 16);
}

__device__ __forceinline__ v8f wmma_bf16(const FragBF& a, const FragBF& b, v8f c) {
    return __builtin_amdgcn_wmma_f32_16x16x32_bf16(
        /*neg_a=*/false, a.v, /*neg_b=*/false, b.v,
        /*c_mod=*/(short)0, c, /*reuse_a=*/false, /*reuse_b=*/false);
}

// ---------------------------------------------------------------------------
// Kernel 1: fp32 -> bf16 convert (query/key/value), 4 elements per thread
// ---------------------------------------------------------------------------
__global__ void cvt_kernel(const float4* __restrict__ q, const float4* __restrict__ k,
                           const float4* __restrict__ v,
                           uint2* __restrict__ oq, uint2* __restrict__ ok,
                           uint2* __restrict__ ov, int n4) {
    int i = blockIdx.x * blockDim.x + threadIdx.x;
    if (i >= n4) return;
    const float4* s = (blockIdx.y == 0) ? q : (blockIdx.y == 1) ? k : v;
    uint2* d       = (blockIdx.y == 0) ? oq : (blockIdx.y == 1) ? ok : ov;
    float4 f = s[i];
    uint2 r;
    r.x = (unsigned)f2bf(f.x) | ((unsigned)f2bf(f.y) << 16);
    r.y = (unsigned)f2bf(f.z) | ((unsigned)f2bf(f.w) << 16);
    d[i] = r;
}

// ---------------------------------------------------------------------------
// Kernel 2: weight transpose + convert:  W fp32 [in][out] -> Wt bf16 [out][in]
// ---------------------------------------------------------------------------
__global__ void wtrans_kernel(const float* __restrict__ W0, const float* __restrict__ W1,
                              const float* __restrict__ W2, const float* __restrict__ W3,
                              unsigned short* __restrict__ O0, unsigned short* __restrict__ O1,
                              unsigned short* __restrict__ O2, unsigned short* __restrict__ O3) {
    __shared__ float tile[32][33];
    const float* W = (blockIdx.z == 0) ? W0 : (blockIdx.z == 1) ? W1
                   : (blockIdx.z == 2) ? W2 : W3;
    unsigned short* O = (blockIdx.z == 0) ? O0 : (blockIdx.z == 1) ? O1
                      : (blockIdx.z == 2) ? O2 : O3;
    int x  = blockIdx.x * 32 + threadIdx.x;
    int y0 = blockIdx.y * 32;
    for (int i = 0; i < 32; i += 8)
        tile[threadIdx.y + i][threadIdx.x] = W[(size_t)(y0 + threadIdx.y + i) * N_D + x];
    __syncthreads();
    int xo  = blockIdx.y * 32 + threadIdx.x;
    int yo0 = blockIdx.x * 32;
    for (int i = 0; i < 32; i += 8)
        O[(size_t)(yo0 + threadIdx.y + i) * N_D + xo] = f2bf(tile[threadIdx.x][threadIdx.y + i]);
}

// ---------------------------------------------------------------------------
// Kernel 3: bf16 GEMM  Y[M=8192][N=1024] = X[M][K=1024] * Wt[N][K]^T + bias
// 128 threads (4 waves), 128x128 block tile, 64x64 wave tile, double-buffered
// async global->LDS staging overlapped with WMMA.
// mode 0: bf16 [B][H][S][DH] | mode 1: bf16 [B][H][DH][S] | mode 2: fp32 [M][N]
// ---------------------------------------------------------------------------
__global__ __launch_bounds__(128) void gemm_bf16_kernel(
    const unsigned short* __restrict__ X,
    const unsigned short* __restrict__ Wt,
    const float* __restrict__ bias,
    unsigned short* __restrict__ outBf,
    float* __restrict__ outF,
    int mode)
{
    const int K = N_D, LD = N_D;
    __shared__ unsigned short As[2][128 * 32];
    __shared__ unsigned short Bs[2][128 * 32];

    const int tid   = threadIdx.x;
    const int lane  = tid & 31;
    const int wave  = tid >> 5;
    const int waveM = wave >> 1;
    const int waveN = wave & 1;
    const int m0 = blockIdx.y * 128;
    const int n0 = blockIdx.x * 128;

    v8f acc[4][4];
    for (int mf = 0; mf < 4; ++mf)
        for (int nf = 0; nf < 4; ++nf)
            for (int e = 0; e < 8; ++e) acc[mf][nf][e] = 0.0f;

    // stage one 128x32 A tile + 128x32 B tile: 8 x b128 per thread
    auto stage = [&](int kk, int buf) {
#if HAVE_ASYNC_COPY
        for (int i = 0; i < 4; ++i) {
            int linear = i * 128 + tid;
            int r = linear >> 2;
            int c = (linear & 3) * 8;
            async_copy_b128(&X [(size_t)(m0 + r) * LD + kk + c], &As[buf][r * 32 + c]);
            async_copy_b128(&Wt[(size_t)(n0 + r) * LD + kk + c], &Bs[buf][r * 32 + c]);
        }
#else
        uint4 ta[4], tb[4];
        for (int i = 0; i < 4; ++i) {
            int linear = i * 128 + tid;
            int r = linear >> 2;
            int c = (linear & 3) * 8;
            ta[i] = *reinterpret_cast<const uint4*>(&X [(size_t)(m0 + r) * LD + kk + c]);
            tb[i] = *reinterpret_cast<const uint4*>(&Wt[(size_t)(n0 + r) * LD + kk + c]);
        }
        for (int i = 0; i < 4; ++i) {
            int linear = i * 128 + tid;
            int r = linear >> 2;
            int c = (linear & 3) * 8;
            *reinterpret_cast<uint4*>(&As[buf][r * 32 + c]) = ta[i];
            *reinterpret_cast<uint4*>(&Bs[buf][r * 32 + c]) = tb[i];
        }
#endif
    };

    const int T = K / 32;            // 32 k-tiles
    stage(0, 0);
    stage(32, 1);

    for (int t = 0; t < T; ++t) {
        if (t + 1 < T) { WAIT_ASYNC(8); } else { WAIT_ASYNC(0); }
        __syncthreads();
        const unsigned short* Ab = As[t & 1];
        const unsigned short* Bb = Bs[t & 1];
        FragBF a[4], b[4];
        for (int mf = 0; mf < 4; ++mf)
            loadA(&Ab[(waveM * 64 + mf * 16) * 32], 32, lane, a[mf]);
        for (int nf = 0; nf < 4; ++nf)
            loadB(&Bb[(waveN * 64 + nf * 16) * 32], 32, lane, b[nf]);
        for (int mf = 0; mf < 4; ++mf)
            for (int nf = 0; nf < 4; ++nf)
                acc[mf][nf] = wmma_bf16(a[mf], b[nf], acc[mf][nf]);
        __syncthreads();
        if (t + 2 < T) stage((t + 2) * 32, t & 1);
    }

    // Epilogue: C frag element (v, lane): row = v + (lane>>4)*8, col = lane&15
    const int li = lane & 15, hi = lane >> 4;
    for (int mf = 0; mf < 4; ++mf) {
        for (int nf = 0; nf < 4; ++nf) {
            int col = n0 + waveN * 64 + nf * 16 + li;
            float bv = bias[col];
            for (int v = 0; v < 8; ++v) {
                int row = m0 + waveM * 64 + mf * 16 + v + hi * 8;
                float val = acc[mf][nf][v] + bv;
                if (mode == 2) {
                    outF[(size_t)row * N_D + col] = val;
                } else {
                    int bb = row >> 11;
                    int s  = row & (N_S - 1);
                    int h  = col >> 6;
                    int dh = col & (N_DH - 1);
                    size_t idx = (mode == 0)
                        ? ((size_t)((bb * N_H + h) * N_S + s)) * N_DH + dh
                        : ((size_t)((bb * N_H + h) * N_DH + dh)) * N_S + s;
                    outBf[idx] = f2bf(val);
                }
            }
        }
    }
}

// ---------------------------------------------------------------------------
// Kernel 4: flash attention.  Q [BH][S][DH], K [BH][S][DH], V^T [BH][DH][S].
// 128 threads (4 waves), 128 query rows, key tiles of 64, double-buffered
// async K/V staging. Online softmax fp32; both GEMMs on WMMA bf16.
// Output: bf16 [B][S][H*DH].
// ---------------------------------------------------------------------------
__global__ __launch_bounds__(128) void attn_kernel(
    const unsigned short* __restrict__ Q,
    const unsigned short* __restrict__ Kp,
    const unsigned short* __restrict__ Vt,
    const unsigned char*  __restrict__ mask,
    unsigned short* __restrict__ Out)
{
    __shared__ unsigned short Qs[128 * 64];        // 16 KB
    __shared__ unsigned short Ks[2][64 * 64];      // 16 KB (row = key)
    __shared__ unsigned short Vs[2][64 * 64];      // 16 KB (row = dh)
    __shared__ unsigned short Ps[4][32 * 64];      // 16 KB (per-wave P tiles)

    const int tid  = threadIdx.x;
    const int lane = tid & 31;
    const int wave = tid >> 5;
    const int li   = lane & 15;
    const int hi   = lane >> 4;

    const int bh = blockIdx.y;
    const int b  = bh >> 4;
    const int h  = bh & (N_H - 1);
    const int q0 = blockIdx.x * 128;

    const unsigned short* Qg = Q  + ((size_t)bh * N_S + q0) * N_DH;
    const unsigned short* Kg = Kp + (size_t)bh * N_S * N_DH;
    const unsigned short* Vg = Vt + (size_t)bh * N_DH * N_S;
    const unsigned char*  mrow = mask + (size_t)b * N_S;

    auto stageKV = [&](int j, int buf) {
#if HAVE_ASYNC_COPY
        for (int i = 0; i < 4; ++i) {
            int linear = i * 128 + tid;       // 0..511
            int r = linear >> 3;              // 0..63
            int c = (linear & 7) * 8;         // 0..56
            async_copy_b128(&Kg[(size_t)(j + r) * 64 + c], &Ks[buf][r * 64 + c]);
            async_copy_b128(&Vg[(size_t)r * N_S + j + c],  &Vs[buf][r * 64 + c]);
        }
#else
        uint4 tk[4], tv[4];
        for (int i = 0; i < 4; ++i) {
            int linear = i * 128 + tid;
            int r = linear >> 3;
            int c = (linear & 7) * 8;
            tk[i] = *reinterpret_cast<const uint4*>(&Kg[(size_t)(j + r) * 64 + c]);
            tv[i] = *reinterpret_cast<const uint4*>(&Vg[(size_t)r * N_S + j + c]);
        }
        for (int i = 0; i < 4; ++i) {
            int linear = i * 128 + tid;
            int r = linear >> 3;
            int c = (linear & 7) * 8;
            *reinterpret_cast<uint4*>(&Ks[buf][r * 64 + c]) = tk[i];
            *reinterpret_cast<uint4*>(&Vs[buf][r * 64 + c]) = tv[i];
        }
#endif
    };

    // stage 128x64 Q tile (8 x b128 per thread)
#if HAVE_ASYNC_COPY
    for (int i = 0; i < 8; ++i) {
        int linear = i * 128 + tid;
        int r = linear >> 3;
        int c = (linear & 7) * 8;
        async_copy_b128(&Qg[(size_t)r * 64 + c], &Qs[r * 64 + c]);
    }
#else
    for (int i = 0; i < 8; ++i) {
        int linear = i * 128 + tid;
        int r = linear >> 3;
        int c = (linear & 7) * 8;
        *reinterpret_cast<uint4*>(&Qs[r * 64 + c]) =
            *reinterpret_cast<const uint4*>(&Qg[(size_t)r * 64 + c]);
    }
#endif

    const int T = N_S / 64;          // 32 key tiles
    stageKV(0, 0);
    stageKV(64, 1);

    WAIT_ASYNC(16);                  // Q's 8 chunks done (kv0+kv1 still in flight)
    __syncthreads();

    FragBF qf[2][2];
    for (int mf = 0; mf < 2; ++mf)
        for (int ks = 0; ks < 2; ++ks)
            loadA(&Qs[(wave * 32 + mf * 16) * 64 + ks * 32], 64, lane, qf[mf][ks]);

    v8f o[2][4];
    float mstat[2][8], lstat[2][8];
    for (int mf = 0; mf < 2; ++mf) {
        for (int df = 0; df < 4; ++df)
            for (int e = 0; e < 8; ++e) o[mf][df][e] = 0.0f;
        for (int v = 0; v < 8; ++v) { mstat[mf][v] = -1e30f; lstat[mf][v] = 0.0f; }
    }
    const float sm_scale = 0.125f;   // 1/sqrt(DH)

    for (int jt = 0; jt < T; ++jt) {
        if (jt + 1 < T) { WAIT_ASYNC(8); } else { WAIT_ASYNC(0); }
        __syncthreads();
        const int j = jt * 64;
        const unsigned short* Kb = Ks[jt & 1];
        const unsigned short* Vb = Vs[jt & 1];

        // scores = Q * K^T
        v8f sc[2][4];
        for (int mf = 0; mf < 2; ++mf)
            for (int nf = 0; nf < 4; ++nf)
                for (int e = 0; e < 8; ++e) sc[mf][nf][e] = 0.0f;
        for (int nf = 0; nf < 4; ++nf) {
            for (int ks = 0; ks < 2; ++ks) {
                FragBF kf;
                loadB(&Kb[(nf * 16) * 64 + ks * 32], 64, lane, kf);
                for (int mf = 0; mf < 2; ++mf)
                    sc[mf][nf] = wmma_bf16(qf[mf][ks], kf, sc[mf][nf]);
            }
        }

        float colm[4];
        for (int nf = 0; nf < 4; ++nf)
            colm[nf] = mrow[j + nf * 16 + li] ? 0.0f : -1e30f;

        // online softmax update
        for (int mf = 0; mf < 2; ++mf) {
            for (int nf = 0; nf < 4; ++nf)
                for (int v = 0; v < 8; ++v)
                    sc[mf][nf][v] = sc[mf][nf][v] * sm_scale + colm[nf];
            for (int v = 0; v < 8; ++v) {
                float m = sc[mf][0][v];
                m = fmaxf(m, sc[mf][1][v]);
                m = fmaxf(m, sc[mf][2][v]);
                m = fmaxf(m, sc[mf][3][v]);
                for (int off = 1; off < 16; off <<= 1)
                    m = fmaxf(m, __shfl_xor(m, off, 32));
                float mnew = fmaxf(mstat[mf][v], m);
                float cf   = __expf(mstat[mf][v] - mnew);
                mstat[mf][v] = mnew;
                float rs = 0.0f;
                for (int nf = 0; nf < 4; ++nf) {
                    float p = __expf(sc[mf][nf][v] - mnew);
                    sc[mf][nf][v] = p;
                    rs += p;
                }
                for (int off = 1; off < 16; off <<= 1)
                    rs += __shfl_xor(rs, off, 32);
                lstat[mf][v] = lstat[mf][v] * cf + rs;
                for (int df = 0; df < 4; ++df)
                    o[mf][df][v] *= cf;
            }
            for (int nf = 0; nf < 4; ++nf)
                for (int v = 0; v < 8; ++v)
                    Ps[wave][(mf * 16 + v + hi * 8) * 64 + nf * 16 + li] =
                        f2bf(sc[mf][nf][v]);
        }

        // O += P * V
        for (int ks = 0; ks < 2; ++ks) {
            FragBF pf[2];
            for (int mf = 0; mf < 2; ++mf)
                loadA(&Ps[wave][(mf * 16) * 64 + ks * 32], 64, lane, pf[mf]);
            for (int df = 0; df < 4; ++df) {
                FragBF vf;
                loadB(&Vb[(df * 16) * 64 + ks * 32], 64, lane, vf);
                for (int mf = 0; mf < 2; ++mf)
                    o[mf][df] = wmma_bf16(pf[mf], vf, o[mf][df]);
            }
        }
        __syncthreads();
        if (jt + 2 < T) stageKV((jt + 2) * 64, jt & 1);
    }

    // normalize and write bf16 [B][S][H*DH]
    for (int mf = 0; mf < 2; ++mf) {
        for (int df = 0; df < 4; ++df) {
            for (int v = 0; v < 8; ++v) {
                int qrow = q0 + wave * 32 + mf * 16 + v + hi * 8;
                int col  = h * N_DH + df * 16 + li;
                float val = o[mf][df][v] / lstat[mf][v];
                Out[((size_t)b * N_S + qrow) * N_D + col] = f2bf(val);
            }
        }
    }
}

// ---------------------------------------------------------------------------
// Host launcher
// ---------------------------------------------------------------------------
extern "C" void kernel_launch(void* const* d_in, const int* in_sizes, int n_in,
                              void* d_out, int out_size, void* d_ws, size_t ws_size,
                              hipStream_t stream) {
    (void)in_sizes; (void)n_in; (void)out_size; (void)ws_size;

    const float* query = (const float*)d_in[0];
    const float* key   = (const float*)d_in[1];
    const float* value = (const float*)d_in[2];
    const unsigned char* mask = (const unsigned char*)d_in[3];
    const float* Wq = (const float*)d_in[4];
    const float* bq = (const float*)d_in[5];
    const float* Wk = (const float*)d_in[6];
    const float* bk = (const float*)d_in[7];
    const float* Wv = (const float*)d_in[8];
    const float* bv = (const float*)d_in[9];
    const float* Wo = (const float*)d_in[10];
    const float* bo = (const float*)d_in[11];
    float* out = (float*)d_out;

    char* ws = (char*)d_ws;
    size_t off = 0;
    auto walloc = [&](size_t bytes) -> void* {
        void* p = ws + off;
        off += (bytes + 255) & ~(size_t)255;
        return p;
    };
    const size_t actB = (size_t)N_BS * N_D * sizeof(unsigned short);  // 16 MB
    const size_t wB   = (size_t)N_D * N_D * sizeof(unsigned short);   //  2 MB

    unsigned short* Xq  = (unsigned short*)walloc(actB);
    unsigned short* Xk  = (unsigned short*)walloc(actB);
    unsigned short* Xv  = (unsigned short*)walloc(actB);
    unsigned short* Wqt = (unsigned short*)walloc(wB);
    unsigned short* Wkt = (unsigned short*)walloc(wB);
    unsigned short* Wvt = (unsigned short*)walloc(wB);
    unsigned short* Wot = (unsigned short*)walloc(wB);
    unsigned short* Qp  = (unsigned short*)walloc(actB);   // [B][H][S][DH]
    unsigned short* Kpj = (unsigned short*)walloc(actB);   // [B][H][S][DH]
    unsigned short* Vtp = (unsigned short*)walloc(actB);   // [B][H][DH][S]
    unsigned short* AO  = (unsigned short*)walloc(actB);   // [B][S][H*DH]

    const int n4 = (N_BS * N_D) / 4;
    dim3 cvtGrid((n4 + 255) / 256, 3);
    cvt_kernel<<<cvtGrid, 256, 0, stream>>>(
        (const float4*)query, (const float4*)key, (const float4*)value,
        (uint2*)Xq, (uint2*)Xk, (uint2*)Xv, n4);

    wtrans_kernel<<<dim3(32, 32, 4), dim3(32, 8), 0, stream>>>(
        Wq, Wk, Wv, Wo, Wqt, Wkt, Wvt, Wot);

    dim3 gGrid(N_D / 128, N_BS / 128);                     // (8, 64)
    gemm_bf16_kernel<<<gGrid, 128, 0, stream>>>(Xq, Wqt, bq, Qp,  nullptr, 0);
    gemm_bf16_kernel<<<gGrid, 128, 0, stream>>>(Xk, Wkt, bk, Kpj, nullptr, 0);
    gemm_bf16_kernel<<<gGrid, 128, 0, stream>>>(Xv, Wvt, bv, Vtp, nullptr, 1);

    attn_kernel<<<dim3(N_S / 128, N_B * N_H), 128, 0, stream>>>(
        Qp, Kpj, Vtp, mask, AO);

    gemm_bf16_kernel<<<gGrid, 128, 0, stream>>>(AO, Wot, bo, nullptr, out, 2);
}
